// GAT_67018669686918
// MI455X (gfx1250) — matrix operand order repaired and checked by reading, not compiled, not run on hardware
//
#include <hip/hip_runtime.h>
#include <hip/hip_bf16.h>

typedef __attribute__((ext_vector_type(16))) __bf16 v16bf;
typedef __attribute__((ext_vector_type(8)))  float  v8f;

#define NN      50000
#define KDIM    128
#define HEADS   8
#define CH      16      // per-head channels (HID == NUM_CLASSES == 16)
#define NEG     0.2f

// ---------- helpers ----------
__device__ __forceinline__ unsigned short f2bf(float f) {
    unsigned int b = __float_as_uint(f);
    b = b + 0x7FFFu + ((b >> 16) & 1u);   // RNE
    return (unsigned short)(b >> 16);
}
// monotone float<->uint encode for atomicMax-based segment max
__device__ __forceinline__ unsigned int fenc(float f) {
    unsigned int b = __float_as_uint(f);
    return b ^ (unsigned int)(((int)b >> 31) | 0x80000000);
}
__device__ __forceinline__ float fdec(unsigned int u) {
    unsigned int b = (u & 0x80000000u) ? (u ^ 0x80000000u) : ~u;
    return __uint_as_float(b);
}

// ---------- elementwise ----------
__global__ void k_f32_to_bf16(const float* __restrict__ in,
                              unsigned short* __restrict__ out, int n) {
    int i = blockIdx.x * blockDim.x + threadIdx.x;
    if (i < n) out[i] = f2bf(in[i]);
}

__global__ void k_relu_bf16(const float* __restrict__ in,
                            unsigned short* __restrict__ out, int n) {
    int i = blockIdx.x * blockDim.x + threadIdx.x;
    if (i < n) { float v = in[i]; out[i] = f2bf(v > 0.f ? v : 0.f); }
}

// init segment max (encoded), segment sum, and bias-seeded accumulator
__global__ void k_init(unsigned int* __restrict__ menc, float* __restrict__ ssum,
                       float* __restrict__ hacc, const float* __restrict__ bias) {
    int i = blockIdx.x * blockDim.x + threadIdx.x;
    if (i < NN * HEADS) { menc[i] = 0u; ssum[i] = 0.f; }
    if (i < NN * KDIM)  hacc[i] = bias[i & (KDIM - 1)];
}

// ---------- dual GEMM: Ol = bf16(X)@Wl, Or = bf16(X)@Wr,  K=N=128 ----------
__global__ __launch_bounds__(256) void k_gemm_dual(
    const unsigned short* __restrict__ X,    // [nrows,128] bf16
    const unsigned short* __restrict__ Wl,   // [128,128] bf16 row-major (k-major)
    const unsigned short* __restrict__ Wr,
    float* __restrict__ Ol, float* __restrict__ Or, int nrows)
{
    // B-fragments pre-swizzled: [mat][kc][nt][lane][elem] -> 32B contiguous per lane
    __shared__ unsigned short sw[2][4][8][32][16];   // 64 KB
    for (int m = 0; m < 2; ++m) {
        const unsigned short* W = m ? Wr : Wl;
        for (int i = threadIdx.x; i < KDIM * KDIM; i += 256) {
            int k = i >> 7, n = i & 127;
            int kc = k >> 5, kk = k & 31, nt = n >> 4, c = n & 15;
            sw[m][kc][nt][((kk >> 4) << 4) | c][kk & 15] = W[i];
        }
    }
    __syncthreads();

    const int wave = threadIdx.x >> 5;
    const int lane = threadIdx.x & 31;
    const int tile = blockIdx.x * 4 + (wave >> 1);
    const int mat  = wave & 1;
    const int r0   = tile * 16;
    const bool valid = (r0 < nrows);
    const int half = lane >> 4;
    const int arow = min(r0 + (lane & 15), nrows - 1);
    const unsigned short* xrow = X + (size_t)arow * KDIM;

    v8f acc[8];
#pragma unroll
    for (int nt = 0; nt < 8; ++nt) acc[nt] = (v8f){0,0,0,0,0,0,0,0};

#pragma unroll
    for (int kc = 0; kc < 4; ++kc) {
        union { uint4 u[2]; v16bf v; } A;
        A.u[0] = *(const uint4*)(xrow + kc * 32 + half * 16);
        A.u[1] = *(const uint4*)(xrow + kc * 32 + half * 16 + 8);
#pragma unroll
        for (int nt = 0; nt < 8; ++nt) {
            union { uint4 u[2]; v16bf v; } B;
            B.u[0] = *(const uint4*)&sw[mat][kc][nt][lane][0];
            B.u[1] = *(const uint4*)&sw[mat][kc][nt][lane][8];
            acc[nt] = __builtin_amdgcn_wmma_f32_16x16x32_bf16(
                false, A.v, false, B.v, (short)0, acc[nt], false, false);
        }
    }
    if (valid) {
        float* O = (mat ? Or : Ol) + (size_t)(r0 + half * 8) * KDIM + (lane & 15);
#pragma unroll
        for (int nt = 0; nt < 8; ++nt)
#pragma unroll
            for (int g = 0; g < 8; ++g)
                O[(size_t)g * KDIM + nt * 16] = acc[nt][g];
    }
}

// ---------- head GEMM: O = bf16(X)@Wlin + blin, 128 -> 16 ----------
__global__ __launch_bounds__(256) void k_gemm_lin(
    const unsigned short* __restrict__ X, const unsigned short* __restrict__ W,
    const float* __restrict__ bias, float* __restrict__ O, int nrows)
{
    __shared__ unsigned short sw[4][32][16];   // 4 KB
    for (int i = threadIdx.x; i < KDIM * 16; i += 256) {
        int k = i >> 4, n = i & 15;
        int kc = k >> 5, kk = k & 31;
        sw[kc][((kk >> 4) << 4) | n][kk & 15] = W[i];
    }
    __syncthreads();

    const int wave = threadIdx.x >> 5, lane = threadIdx.x & 31;
    const int tile = blockIdx.x * 8 + wave;
    const int r0 = tile * 16;
    const bool valid = (r0 < nrows);
    const int half = lane >> 4;
    const int arow = min(r0 + (lane & 15), nrows - 1);
    const unsigned short* xrow = X + (size_t)arow * KDIM;

    v8f acc = (v8f){0,0,0,0,0,0,0,0};
#pragma unroll
    for (int kc = 0; kc < 4; ++kc) {
        union { uint4 u[2]; v16bf v; } A;
        A.u[0] = *(const uint4*)(xrow + kc * 32 + half * 16);
        A.u[1] = *(const uint4*)(xrow + kc * 32 + half * 16 + 8);
        union { uint4 u[2]; v16bf v; } B;
        B.u[0] = *(const uint4*)&sw[kc][lane][0];
        B.u[1] = *(const uint4*)&sw[kc][lane][8];
        acc = __builtin_amdgcn_wmma_f32_16x16x32_bf16(
            false, A.v, false, B.v, (short)0, acc, false, false);
    }
    if (valid) {
        const int col = lane & 15;
        const float b = bias[col];
        float* o = O + (size_t)(r0 + half * 8) * 16 + col;
#pragma unroll
        for (int g = 0; g < 8; ++g) o[(size_t)g * 16] = acc[g] + b;
    }
}

// ---------- edge passes (thread = one (edge, head)) ----------
__global__ void k_edge_logits(const int* __restrict__ src, const int* __restrict__ dst,
                              int e0, int etot, const float* __restrict__ xl,
                              const float* __restrict__ xr, const float* __restrict__ att,
                              float* __restrict__ ebuf, unsigned int* __restrict__ menc)
{
    int t = blockIdx.x * blockDim.x + threadIdx.x;
    if (t >= etot * HEADS) return;
    int e = t >> 3, h = t & 7;
    int si = (e < e0) ? src[e] : (e - e0);
    int di = (e < e0) ? dst[e] : (e - e0);
    const float4* pl = (const float4*)(xl + (size_t)si * KDIM + h * CH);
    const float4* pr = (const float4*)(xr + (size_t)di * KDIM + h * CH);
    const float4* pa = (const float4*)(att + h * CH);
    float logit = 0.f;
#pragma unroll
    for (int q = 0; q < 4; ++q) {
        float4 a = pl[q], b = pr[q], w = pa[q];
        float v;
        v = a.x + b.x; v = v > 0.f ? v : NEG * v; logit += w.x * v;
        v = a.y + b.y; v = v > 0.f ? v : NEG * v; logit += w.y * v;
        v = a.z + b.z; v = v > 0.f ? v : NEG * v; logit += w.z * v;
        v = a.w + b.w; v = v > 0.f ? v : NEG * v; logit += w.w * v;
    }
    ebuf[t] = logit;
    atomicMax(&menc[(size_t)di * HEADS + h], fenc(logit));
}

__global__ void k_edge_norm(const int* __restrict__ src, const int* __restrict__ dst,
                            int e0, int etot, const unsigned int* __restrict__ menc,
                            float* __restrict__ ebuf, float* __restrict__ ssum)
{
    int t = blockIdx.x * blockDim.x + threadIdx.x;
    if (t >= etot * HEADS) return;
    int e = t >> 3, h = t & 7;
    int di = (e < e0) ? dst[e] : (e - e0);
    (void)src;
    float m = fdec(menc[(size_t)di * HEADS + h]);
    float a = __expf(ebuf[t] - m);
    ebuf[t] = a;
    atomicAdd(&ssum[(size_t)di * HEADS + h], a);
}

__global__ void k_edge_aggr(const int* __restrict__ src, const int* __restrict__ dst,
                            int e0, int etot, const float* __restrict__ ebuf,
                            const float* __restrict__ ssum, const float* __restrict__ xl,
                            float* __restrict__ hacc)
{
    int t = blockIdx.x * blockDim.x + threadIdx.x;
    if (t >= etot * HEADS) return;
    int e = t >> 3, h = t & 7;
    int si = (e < e0) ? src[e] : (e - e0);
    int di = (e < e0) ? dst[e] : (e - e0);
    float alpha = ebuf[t] / (ssum[(size_t)di * HEADS + h] + 1e-16f);
    const float4* pl = (const float4*)(xl + (size_t)si * KDIM + h * CH);
    float* po = hacc + (size_t)di * KDIM + h * CH;
#pragma unroll
    for (int q = 0; q < 4; ++q) {
        float4 v = pl[q];
        atomicAdd(po + 4 * q + 0, alpha * v.x);
        atomicAdd(po + 4 * q + 1, alpha * v.y);
        atomicAdd(po + 4 * q + 2, alpha * v.z);
        atomicAdd(po + 4 * q + 3, alpha * v.w);
    }
}

__global__ void k_logsoftmax(const float* __restrict__ in, float* __restrict__ out, int n) {
    int r = blockIdx.x * blockDim.x + threadIdx.x;
    if (r >= n) return;
    const float* p = in + (size_t)r * 16;
    float mx = p[0];
#pragma unroll
    for (int c = 1; c < 16; ++c) mx = fmaxf(mx, p[c]);
    float s = 0.f;
#pragma unroll
    for (int c = 0; c < 16; ++c) s += __expf(p[c] - mx);
    float l = __logf(s) + mx;
    float* o = out + (size_t)r * 16;
#pragma unroll
    for (int c = 0; c < 16; ++c) o[c] = p[c] - l;
}

// ---------- launch ----------
extern "C" void kernel_launch(void* const* d_in, const int* in_sizes, int n_in,
                              void* d_out, int out_size, void* d_ws, size_t ws_size,
                              hipStream_t stream) {
    (void)n_in; (void)out_size; (void)ws_size;
    const float* x    = (const float*)d_in[0];
    const int*   edge = (const int*)d_in[1];
    const float* Wl1  = (const float*)d_in[2];
    const float* Wr1  = (const float*)d_in[3];
    const float* att1 = (const float*)d_in[4];
    const float* b1   = (const float*)d_in[5];
    const float* Wl2  = (const float*)d_in[6];
    const float* Wr2  = (const float*)d_in[7];
    const float* att2 = (const float*)d_in[8];
    const float* b2   = (const float*)d_in[9];
    const float* Wlin = (const float*)d_in[10];
    const float* blin = (const float*)d_in[11];
    const int E0 = in_sizes[1] / 2;
    const int ET = E0 + NN;
    const int* srcA = edge;
    const int* dstA = edge + E0;

    char* w = (char*)d_ws; size_t off = 0;
    auto bump = [&](size_t bytes) { void* p = w + off; off = (off + bytes + 255) & ~(size_t)255; return p; };
    unsigned short* xbf   = (unsigned short*)bump((size_t)NN * KDIM * 2);
    unsigned short* wl1b  = (unsigned short*)bump(KDIM * KDIM * 2);
    unsigned short* wr1b  = (unsigned short*)bump(KDIM * KDIM * 2);
    unsigned short* wl2b  = (unsigned short*)bump(KDIM * KDIM * 2);
    unsigned short* wr2b  = (unsigned short*)bump(KDIM * KDIM * 2);
    unsigned short* wlinb = (unsigned short*)bump(KDIM * 16 * 2);
    float*        xl   = (float*)bump((size_t)NN * KDIM * 4);
    float*        xr   = (float*)bump((size_t)NN * KDIM * 4);
    unsigned int* menc = (unsigned int*)bump((size_t)NN * HEADS * 4);
    float*        ssum = (float*)bump((size_t)NN * HEADS * 4);
    float*        hacc = (float*)bump((size_t)NN * KDIM * 4);
    float*        ebuf = (float*)bump((size_t)ET * HEADS * 4);
    float*        lbuf = ebuf;   // reuse: edge buffer is free by head-GEMM time

    const int BT = 256;
    const int gElem  = (NN * KDIM + BT - 1) / BT;     // 6.4M elems
    const int gEdge  = (ET * HEADS + BT - 1) / BT;    // 13.2M items
    const int gGemm  = (NN / 16 + 3) / 4;             // 782 (3125 tiles / 4 per block)
    const int gLin   = (NN / 16 + 7) / 8;             // 391

    // precision conversions
    k_f32_to_bf16<<<gElem, BT, 0, stream>>>(x, xbf, NN * KDIM);
    k_f32_to_bf16<<<64, BT, 0, stream>>>(Wl1, wl1b, KDIM * KDIM);
    k_f32_to_bf16<<<64, BT, 0, stream>>>(Wr1, wr1b, KDIM * KDIM);
    k_f32_to_bf16<<<64, BT, 0, stream>>>(Wl2, wl2b, KDIM * KDIM);
    k_f32_to_bf16<<<64, BT, 0, stream>>>(Wr2, wr2b, KDIM * KDIM);
    k_f32_to_bf16<<<8, BT, 0, stream>>>(Wlin, wlinb, KDIM * 16);

    // ---- conv1 ----
    k_gemm_dual<<<gGemm, BT, 0, stream>>>(xbf, wl1b, wr1b, xl, xr, NN);
    k_init<<<gElem, BT, 0, stream>>>(menc, ssum, hacc, b1);
    k_edge_logits<<<gEdge, BT, 0, stream>>>(srcA, dstA, E0, ET, xl, xr, att1, ebuf, menc);
    k_edge_norm<<<gEdge, BT, 0, stream>>>(srcA, dstA, E0, ET, menc, ebuf, ssum);
    k_edge_aggr<<<gEdge, BT, 0, stream>>>(srcA, dstA, E0, ET, ebuf, ssum, xl, hacc);
    k_relu_bf16<<<gElem, BT, 0, stream>>>(hacc, xbf, NN * KDIM);

    // ---- conv2 ----
    k_gemm_dual<<<gGemm, BT, 0, stream>>>(xbf, wl2b, wr2b, xl, xr, NN);
    k_init<<<gElem, BT, 0, stream>>>(menc, ssum, hacc, b2);
    k_edge_logits<<<gEdge, BT, 0, stream>>>(srcA, dstA, E0, ET, xl, xr, att2, ebuf, menc);
    k_edge_norm<<<gEdge, BT, 0, stream>>>(srcA, dstA, E0, ET, menc, ebuf, ssum);
    k_edge_aggr<<<gEdge, BT, 0, stream>>>(srcA, dstA, E0, ET, ebuf, ssum, xl, hacc);
    k_relu_bf16<<<gElem, BT, 0, stream>>>(hacc, xbf, NN * KDIM);

    // ---- head + log-softmax ----
    k_gemm_lin<<<gLin, BT, 0, stream>>>(xbf, wlinb, blin, lbuf, NN);
    k_logsoftmax<<<(NN + BT - 1) / BT, BT, 0, stream>>>(lbuf, (float*)d_out, NN);
}
